// Agent_65189013618882
// MI455X (gfx1250) — matrix operand order repaired and checked by reading, not compiled, not run on hardware
//
#include <hip/hip_runtime.h>

// Problem dims
#define T_    512
#define B_    128
#define IN_   544
#define H_    512
#define A_    18
#define NOUT_ 19   // A+1

typedef unsigned short u16;
typedef __attribute__((ext_vector_type(16))) __bf16 v16bf;
typedef __attribute__((ext_vector_type(8)))  float  v8f;

static __device__ __forceinline__ u16 f2bf(float f) {
  union { float f; unsigned u; } v; v.f = f;
  unsigned r = v.u + 0x7fffu + ((v.u >> 16) & 1u);   // round-to-nearest-even
  return (u16)(r >> 16);
}
static __device__ __forceinline__ float sigm(float x) {
  return 1.0f / (1.0f + __expf(-x));
}
// Build a 16-element bf16 fragment from two 16-byte chunks (LDS or global).
static __device__ __forceinline__ v16bf load_frag(const u16* p0, const u16* p1) {
  union { uint4 u[2]; v16bf v; } f;
  f.u[0] = *(const uint4*)p0;
  f.u[1] = *(const uint4*)p1;
  return f.v;
}
#define WMMA_BF16(a, b, c) \
  __builtin_amdgcn_wmma_f32_16x16x32_bf16(false, (a), false, (b), (short)0, (c), false, false)

// ---------------------------------------------------------------------------
// Fused LSTM step for one layer: gates = A1 @ W1^T + A2 @ W2^T + bsum,
// then i/f/g/o nonlinearities + cell update, with done-mask on recurrent state.
//
// Blocking: each block owns a 32-batch-row slab x 128 h-cols. Each wave owns
// two 16x16 M-tiles (8 f32 accumulators: 2 M x 4 gates) so every weight (B)
// fragment fetched from L2 feeds 2 WMMAs -> halves per-step L2 weight traffic
// vs 1-M-tile blocking. Staging is two-phase (input proj, then recurrent proj)
// reusing one 32xK bf16 LDS buffer.
// Grid: 16 blocks (4 mSlabs x 4 hGroups), 256 threads (8 waves).
// ---------------------------------------------------------------------------
__global__ __launch_bounds__(256) void k_lstm_step(
    const float* __restrict__ xin, int ldx, int K1,      // layer input (f32)
    const u16*   __restrict__ W1,                        // [4H, K1] bf16
    const u16*   __restrict__ W2,                        // [4H, H ] bf16
    const float* __restrict__ hprev,                     // [B, H] recurrent h (read)
    float*       __restrict__ cbuf,                      // [B, H] cell state (in/out)
    float*       __restrict__ hout,                      // [B, H] new h (write)
    const float* __restrict__ bsum,                      // [4H] b_ih + b_hh
    const int*   __restrict__ done_t,                    // [B]
    u16*         __restrict__ hid_bf)                    // bf16 hidden out (layer 1) or null
{
  __shared__ __align__(16) u16 sA[32 * IN_];   // 34.8 KB, reused for both phases
  const int tid    = threadIdx.x;
  const int mSlab  = blockIdx.x >> 2;      // 0..3  (32 batch rows)
  const int hGroup = blockIdx.x & 3;       // 0..3  (128 h cols)
  const int wave   = tid >> 5;
  const int lane   = tid & 31;
  const int hTile  = hGroup * 128 + wave * 16;
  const int mBase  = mSlab * 32;

  const int m = lane & 15, half = lane >> 4;
  const int ncol = hTile + m;              // this lane's h column
  v8f aI0 = {}, aF0 = {}, aG0 = {}, aO0 = {};
  v8f aI1 = {}, aF1 = {}, aG1 = {}, aO1 = {};

  // ---- Phase 1: input projection  (A1 = layer input, no mask) ----
  for (int idx = tid; idx < 32 * K1; idx += 256) {
    int r = idx / K1, k = idx - r * K1;
    sA[idx] = f2bf(xin[(size_t)(mBase + r) * ldx + k]);
  }
  __syncthreads();
  {
    const u16* sa0 = sA + m * K1;           // M-tile 0 (rows 0..15 of slab)
    const u16* sa1 = sA + (16 + m) * K1;    // M-tile 1 (rows 16..31)
    const size_t gs = (size_t)H_ * K1;
    for (int kc = 0; kc < K1; kc += 32) {
      v16bf a0 = load_frag(sa0 + kc + half * 8, sa0 + kc + 16 + half * 8);
      v16bf a1 = load_frag(sa1 + kc + half * 8, sa1 + kc + 16 + half * 8);
      const u16* w = W1 + (size_t)ncol * K1 + kc + half * 8;
      v16bf bI = load_frag(w,          w + 16);
      v16bf bF = load_frag(w + gs,     w + gs + 16);
      v16bf bG = load_frag(w + 2 * gs, w + 2 * gs + 16);
      v16bf bO = load_frag(w + 3 * gs, w + 3 * gs + 16);
      aI0 = WMMA_BF16(a0, bI, aI0);  aI1 = WMMA_BF16(a1, bI, aI1);
      aF0 = WMMA_BF16(a0, bF, aF0);  aF1 = WMMA_BF16(a1, bF, aF1);
      aG0 = WMMA_BF16(a0, bG, aG0);  aG1 = WMMA_BF16(a1, bG, aG1);
      aO0 = WMMA_BF16(a0, bO, aO0);  aO1 = WMMA_BF16(a1, bO, aO1);
    }
  }
  __syncthreads();

  // ---- Phase 2: recurrent projection  (A2 = h_prev * (1-done)) ----
  for (int idx = tid; idx < 32 * H_; idx += 256) {
    int r = idx >> 9, k = idx & (H_ - 1);
    int gm = mBase + r;
    float mask = 1.0f - (float)done_t[gm];
    sA[idx] = f2bf(hprev[(size_t)gm * H_ + k] * mask);
  }
  __syncthreads();
  {
    const u16* sa0 = sA + m * H_;
    const u16* sa1 = sA + (16 + m) * H_;
    const size_t gs = (size_t)H_ * H_;
    for (int kc = 0; kc < H_; kc += 32) {
      v16bf a0 = load_frag(sa0 + kc + half * 8, sa0 + kc + 16 + half * 8);
      v16bf a1 = load_frag(sa1 + kc + half * 8, sa1 + kc + 16 + half * 8);
      const u16* w = W2 + (size_t)ncol * H_ + kc + half * 8;
      v16bf bI = load_frag(w,          w + 16);
      v16bf bF = load_frag(w + gs,     w + gs + 16);
      v16bf bG = load_frag(w + 2 * gs, w + 2 * gs + 16);
      v16bf bO = load_frag(w + 3 * gs, w + 3 * gs + 16);
      aI0 = WMMA_BF16(a0, bI, aI0);  aI1 = WMMA_BF16(a1, bI, aI1);
      aF0 = WMMA_BF16(a0, bF, aF0);  aF1 = WMMA_BF16(a1, bF, aF1);
      aG0 = WMMA_BF16(a0, bG, aG0);  aG1 = WMMA_BF16(a1, bG, aG1);
      aO0 = WMMA_BF16(a0, bO, aO0);  aO1 = WMMA_BF16(a1, bO, aO1);
    }
  }

  // ---- Fused LSTM cell epilogue ----
  // C/D layout: VGPR r -> row r + 8*half, col = lane&15.
  const float bI = bsum[ncol],          bF = bsum[H_ + ncol],
              bG = bsum[2 * H_ + ncol], bO = bsum[3 * H_ + ncol];
#pragma unroll
  for (int mt = 0; mt < 2; ++mt) {
    const v8f& cI = mt ? aI1 : aI0;
    const v8f& cF = mt ? aF1 : aF0;
    const v8f& cG = mt ? aG1 : aG0;
    const v8f& cO = mt ? aO1 : aO0;
#pragma unroll
    for (int r = 0; r < 8; ++r) {
      int gm = mBase + mt * 16 + r + 8 * half;
      float mask = 1.0f - (float)done_t[gm];
      float iv = sigm(cI[r] + bI);
      float fv = sigm(cF[r] + bF);
      float gv = tanhf(cG[r] + bG);
      float ov = sigm(cO[r] + bO);
      size_t off = (size_t)gm * H_ + ncol;
      float cn = fv * (cbuf[off] * mask) + iv * gv;
      cbuf[off] = cn;
      float hn = ov * tanhf(cn);
      hout[off] = hn;
      if (hid_bf) hid_bf[off] = f2bf(hn);
    }
  }
}

// ---------------------------------------------------------------------------
// Heads: out[T*B, 19] = hidden(bf16) @ Whead^T + bhead. Whead padded to 32 rows.
// Grid: 512 blocks x 8 waves; each wave: one 16-row M tile, both N tiles.
// ---------------------------------------------------------------------------
__global__ __launch_bounds__(256) void k_head(
    const u16* __restrict__ hid, const u16* __restrict__ Whead,
    const float* __restrict__ bhead, float* __restrict__ out)
{
  const int wave = threadIdx.x >> 5, lane = threadIdx.x & 31;
  const int mTile = blockIdx.x * 8 + wave;        // 0..4095
  const int m = lane & 15, half = lane >> 4;
  v8f acc0 = {}, acc1 = {};
  const u16* arow = hid + (size_t)(mTile * 16 + m) * H_;
  const u16* w0 = Whead + (size_t)m * H_;
  const u16* w1 = Whead + (size_t)(16 + m) * H_;
  for (int kc = 0; kc < H_; kc += 32) {
    v16bf a = load_frag(arow + kc + half * 8, arow + kc + 16 + half * 8);
    const u16* p0 = w0 + kc + half * 8;
    const u16* p1 = w1 + kc + half * 8;
    acc0 = WMMA_BF16(a, load_frag(p0, p0 + 16), acc0);
    acc1 = WMMA_BF16(a, load_frag(p1, p1 + 16), acc1);
  }
#pragma unroll
  for (int r = 0; r < 8; ++r) {
    int row = mTile * 16 + r + 8 * half;
    out[(size_t)row * NOUT_ + m] = acc0[r] + bhead[m];
    int c1 = 16 + m;
    if (c1 < NOUT_) out[(size_t)row * NOUT_ + c1] = acc1[r] + bhead[c1];
  }
}

// ------------------------- small utility kernels ---------------------------
__global__ void k_f32_to_bf16(const float* __restrict__ s, u16* __restrict__ d, int n) {
  int i = blockIdx.x * blockDim.x + threadIdx.x;
  if (i < n) d[i] = f2bf(s[i]);
}
__global__ void k_bias_sum(const float* a, const float* b, float* o, int n) {
  int i = blockIdx.x * blockDim.x + threadIdx.x;
  if (i < n) o[i] = a[i] + b[i];
}
__global__ void k_build_head(const float* Wa, const float* Wc, const float* ba,
                             const float* bc, u16* Whead, float* bhead) {
  int idx = blockIdx.x * blockDim.x + threadIdx.x;   // 32*512 threads
  if (idx >= 32 * H_) return;
  int n = idx >> 9, k = idx & (H_ - 1);
  float v = 0.f;
  if (n < A_) v = Wa[n * H_ + k];
  else if (n == A_) v = Wc[k];
  Whead[idx] = f2bf(v);
  if (idx < 32) {
    float bv = 0.f;
    if (idx < A_) bv = ba[idx];
    else if (idx == A_) bv = bc[0];
    bhead[idx] = bv;
  }
}
__global__ void k_init_state(const float* h0, const float* c0,
                             float* h0b, float* h1b, float* c0b, float* c1b) {
  int i = blockIdx.x * blockDim.x + threadIdx.x;     // 65536 threads
  if (i >= B_ * H_) return;
  h0b[i] = h0[i];
  h1b[i] = h0[B_ * H_ + i];
  c0b[i] = c0[i];
  c1b[i] = c0[B_ * H_ + i];
}
__global__ void k_final_state(const float* h0f, const float* h1f,
                              const float* c0f, const float* c1f,
                              float* out, size_t htOff, size_t ctOff) {
  int i = blockIdx.x * blockDim.x + threadIdx.x;
  if (i >= B_ * H_) return;
  out[htOff + i]            = h0f[i];
  out[htOff + B_ * H_ + i]  = h1f[i];
  out[ctOff + i]            = c0f[i];
  out[ctOff + B_ * H_ + i]  = c1f[i];
}

// ---------------------------------------------------------------------------
extern "C" void kernel_launch(void* const* d_in, const int* in_sizes, int n_in,
                              void* d_out, int out_size, void* d_ws, size_t ws_size,
                              hipStream_t stream) {
  (void)in_sizes; (void)n_in; (void)out_size; (void)ws_size;
  const float* x       = (const float*)d_in[0];
  const int*   done    = (const int*)  d_in[1];
  const float* h0      = (const float*)d_in[2];
  const float* c0      = (const float*)d_in[3];
  const float* W_ih0   = (const float*)d_in[4];
  const float* W_hh0   = (const float*)d_in[5];
  const float* b_ih0   = (const float*)d_in[6];
  const float* b_hh0   = (const float*)d_in[7];
  const float* W_ih1   = (const float*)d_in[8];
  const float* W_hh1   = (const float*)d_in[9];
  const float* b_ih1   = (const float*)d_in[10];
  const float* b_hh1   = (const float*)d_in[11];
  const float* W_actor = (const float*)d_in[12];
  const float* b_actor = (const float*)d_in[13];
  const float* W_crit  = (const float*)d_in[14];
  const float* b_crit  = (const float*)d_in[15];
  float* out = (float*)d_out;

  // Workspace layout (256B aligned slabs)
  char* ws = (char*)d_ws;
  size_t o = 0;
  auto slab = [&](size_t bytes) { size_t r = o; o += (bytes + 255) & ~(size_t)255; return r; };
  u16*   Wih0bf = (u16*)  (ws + slab((size_t)4 * H_ * IN_ * 2));
  u16*   Whh0bf = (u16*)  (ws + slab((size_t)4 * H_ * H_  * 2));
  u16*   Wih1bf = (u16*)  (ws + slab((size_t)4 * H_ * H_  * 2));
  u16*   Whh1bf = (u16*)  (ws + slab((size_t)4 * H_ * H_  * 2));
  u16*   Whead  = (u16*)  (ws + slab((size_t)32 * H_ * 2));
  float* bsum0  = (float*)(ws + slab((size_t)4 * H_ * 4));
  float* bsum1  = (float*)(ws + slab((size_t)4 * H_ * 4));
  float* bhead  = (float*)(ws + slab(32 * 4));
  float* h0buf  = (float*)(ws + slab((size_t)2 * B_ * H_ * 4));  // ping/pong
  float* h1buf  = (float*)(ws + slab((size_t)2 * B_ * H_ * 4));
  float* c0buf  = (float*)(ws + slab((size_t)B_ * H_ * 4));
  float* c1buf  = (float*)(ws + slab((size_t)B_ * H_ * 4));
  u16*   hidbf  = (u16*)  (ws + slab((size_t)T_ * B_ * H_ * 2));

  // 1) Pre-pass: bf16 weights, bias sums, head pack, state init.
  int nw;
  nw = 4 * H_ * IN_; k_f32_to_bf16<<<(nw + 255) / 256, 256, 0, stream>>>(W_ih0, Wih0bf, nw);
  nw = 4 * H_ * H_;  k_f32_to_bf16<<<(nw + 255) / 256, 256, 0, stream>>>(W_hh0, Whh0bf, nw);
  nw = 4 * H_ * H_;  k_f32_to_bf16<<<(nw + 255) / 256, 256, 0, stream>>>(W_ih1, Wih1bf, nw);
  nw = 4 * H_ * H_;  k_f32_to_bf16<<<(nw + 255) / 256, 256, 0, stream>>>(W_hh1, Whh1bf, nw);
  k_bias_sum<<<(4 * H_ + 255) / 256, 256, 0, stream>>>(b_ih0, b_hh0, bsum0, 4 * H_);
  k_bias_sum<<<(4 * H_ + 255) / 256, 256, 0, stream>>>(b_ih1, b_hh1, bsum1, 4 * H_);
  k_build_head<<<(32 * H_ + 255) / 256, 256, 0, stream>>>(W_actor, W_crit, b_actor, b_crit, Whead, bhead);
  k_init_state<<<(B_ * H_ + 255) / 256, 256, 0, stream>>>(h0, c0, h0buf, h1buf, c0buf, c1buf);

  // 2) Sequential scan: 2 fused WMMA step-kernels per timestep.
  int wr = 0;
  for (int t = 0; t < T_; ++t) {
    int rd = t & 1; wr = rd ^ 1;
    const float* xt = x + (size_t)t * B_ * IN_;
    const int* dt = done + (size_t)t * B_;
    // layer 0: A1 = x_t, A2 = h0_prev
    k_lstm_step<<<16, 256, 0, stream>>>(xt, IN_, IN_, Wih0bf, Whh0bf,
        h0buf + (size_t)rd * B_ * H_, c0buf, h0buf + (size_t)wr * B_ * H_,
        bsum0, dt, (u16*)nullptr);
    // layer 1: A1 = h0_new, A2 = h1_prev; writes bf16 hidden for the heads
    k_lstm_step<<<16, 256, 0, stream>>>(h0buf + (size_t)wr * B_ * H_, H_, H_,
        Wih1bf, Whh1bf,
        h1buf + (size_t)rd * B_ * H_, c1buf, h1buf + (size_t)wr * B_ * H_,
        bsum1, dt, hidbf + (size_t)t * B_ * H_);
  }

  // 3) Heads over all T*B rows (parallel WMMA GEMM), then final state copy.
  k_head<<<(T_ * B_ / 16) / 8, 256, 0, stream>>>(hidbf, Whead, bhead, out);
  const size_t htOff = (size_t)T_ * B_ * NOUT_;
  const size_t ctOff = htOff + (size_t)2 * B_ * H_;
  k_final_state<<<(B_ * H_ + 255) / 256, 256, 0, stream>>>(
      h0buf + (size_t)wr * B_ * H_, h1buf + (size_t)wr * B_ * H_,
      c0buf, c1buf, out, htOff, ctOff);
}
